// DynamicPooling_23656679866709
// MI455X (gfx1250) — matrix operand order repaired and checked by reading, not compiled, not run on hardware
//
#include <hip/hip_runtime.h>
#include <hip/hip_bf16.h>

// Problem constants (from reference)
#define B_   16
#define T_   8192
#define C_   1024
#define L_   512
#define TEMP_ 50.0f
#define LOG2E 1.44269504f

typedef __bf16 bf16_t;
typedef __attribute__((ext_vector_type(16))) __bf16        v16bf;
typedef __attribute__((ext_vector_type(8)))  float         v8f;
typedef __attribute__((ext_vector_type(4)))  float         f32x4;
typedef __attribute__((ext_vector_type(4)))  unsigned int  u32x4;

// ---------------------------------------------------------------------------
// Kernel 1: scores[b,t] = sigmoid(dot(x[b,t,:], w) + bias)
// One wave (32 lanes) per row; w cached in LDS; float4 global loads.
// ---------------------------------------------------------------------------
__global__ __launch_bounds__(256) void score_kernel(
    const float* __restrict__ x, const float* __restrict__ w,
    const float* __restrict__ bias, float* __restrict__ scores)
{
  __shared__ alignas(16) float sw[C_];
  for (int i = threadIdx.x; i < C_; i += 256) sw[i] = w[i];
  __syncthreads();

  const int wave = threadIdx.x >> 5;
  const int lane = threadIdx.x & 31;
  const size_t row = (size_t)blockIdx.x * 8 + wave;
  const float* xr = x + row * C_;

  float s = 0.f;
#pragma unroll
  for (int j = 0; j < 8; ++j) {
    const int k4 = j * 32 + lane;                 // coalesced: 512B per wave
    const f32x4 xv = *reinterpret_cast<const f32x4*>(xr + (size_t)k4 * 4);
    const f32x4 wv = *reinterpret_cast<const f32x4*>(&sw[k4 * 4]);
    s += xv.x * wv.x + xv.y * wv.y + xv.z * wv.z + xv.w * wv.w;
  }
#pragma unroll
  for (int off = 16; off > 0; off >>= 1) s += __shfl_down(s, off, 32);

  if (lane == 0) {
    const float z = s + bias[0];
    const float e = __builtin_amdgcn_exp2f(-z * LOG2E);     // v_exp_f32
    scores[row] = __builtin_amdgcn_rcpf(1.f + e);           // v_rcp_f32
  }
}

// ---------------------------------------------------------------------------
// Kernel 2: in-place normalized cumsum over T per batch. One 1024-thread
// block per batch; 8 elems/thread + Hillis-Steele LDS scan.
// ---------------------------------------------------------------------------
__global__ __launch_bounds__(1024) void cumsum_kernel(float* __restrict__ nc)
{
  __shared__ float part[1024];
  float* p = nc + (size_t)blockIdx.x * T_;
  const int tid = threadIdx.x;

  float v[8]; float s = 0.f;
#pragma unroll
  for (int j = 0; j < 8; ++j) { v[j] = p[tid * 8 + j]; s += v[j]; }
  part[tid] = s;
  __syncthreads();

  for (int off = 1; off < 1024; off <<= 1) {
    const float t = (tid >= off) ? part[tid - off] : 0.f;
    __syncthreads();
    part[tid] += t;
    __syncthreads();
  }

  const float total  = part[1023];
  const float prefix = (tid > 0) ? part[tid - 1] : 0.f;
  const float inv    = __builtin_amdgcn_rcpf(total + 1e-8f);

  float run = prefix;
#pragma unroll
  for (int j = 0; j < 8; ++j) { run += v[j]; p[tid * 8 + j] = run * inv; }
}

// ---------------------------------------------------------------------------
// Kernel 3: inv_denom[b,l] = 1 / (sum_t membership(b,t,l) + 1e-8)
// Telescoped sigmoid: sig(50(nc - l/L)) = 1/(1 + E*K_l), E = exp(-50*nc).
// ---------------------------------------------------------------------------
__global__ __launch_bounds__(256) void denom_kernel(
    const float* __restrict__ nc, float* __restrict__ inv_denom)
{
  const int b = blockIdx.x >> 9;          // / L_
  const int l = blockIdx.x & (L_ - 1);
  const float* p = nc + (size_t)b * T_;

  const float klo = __builtin_amdgcn_exp2f((TEMP_ * (float)l       / L_) * LOG2E);
  const float khi = __builtin_amdgcn_exp2f((TEMP_ * (float)(l + 1) / L_) * LOG2E);

  float s = 0.f;
  for (int t = threadIdx.x; t < T_; t += 256) {
    const float E = __builtin_amdgcn_exp2f(-TEMP_ * p[t] * LOG2E);
    s += __builtin_amdgcn_rcpf(1.f + E * klo) - __builtin_amdgcn_rcpf(1.f + E * khi);
  }

  __shared__ float red[256];
  red[threadIdx.x] = s;
  __syncthreads();
  for (int off = 128; off > 0; off >>= 1) {
    if (threadIdx.x < off) red[threadIdx.x] += red[threadIdx.x + off];
    __syncthreads();
  }
  if (threadIdx.x == 0)
    inv_denom[blockIdx.x] = __builtin_amdgcn_rcpf(red[0] + 1e-8f);
}

// ---------------------------------------------------------------------------
// Kernel 4: pooled[b,l,c] = sum_t m_norm(b,t,l) * x[b,t,c]  via bf16 WMMA.
// Block tile: 64 l x 128 c, K-step 32 over T. 8 waves; each wave owns a
// 16 l x 64 c strip = 4 accumulators of v_wmma_f32_16x16x32_bf16.
// ---------------------------------------------------------------------------
__global__ __launch_bounds__(256) void pool_gemm_kernel(
    const float* __restrict__ x, const float* __restrict__ nc,
    const float* __restrict__ inv_denom, float* __restrict__ out)
{
  __shared__ alignas(16) bf16_t smemA[64 * 32];    // [l][t]  membership, bf16
  __shared__ alignas(16) bf16_t smemB[128 * 32];   // [c][t]  x transposed, bf16

  const int tid  = threadIdx.x;
  const int lane = tid & 31;
  const int wave = tid >> 5;
  const int half = lane >> 4;
  const int n16  = lane & 15;
  const int wr   = wave & 3;     // wave's l-strip (0..3)
  const int wc   = wave >> 2;    // wave's c-half  (0..1)

  const int cblock = blockIdx.x * 128;
  const int lblock = blockIdx.y * 64;
  const int b      = blockIdx.z;

  const float* nc_b = nc + (size_t)b * T_;
  const float* xb   = x  + (size_t)b * T_ * C_;

  // Stage-A work assignment: thread handles t = tA, l = lg*8 + i (i=0..7).
  const int tA = tid & 31;
  const int lg = tid >> 5;
  float K[9], idv[8];
#pragma unroll
  for (int i = 0; i < 9; ++i) {
    const int lglob = lblock + lg * 8 + i;
    K[i] = __builtin_amdgcn_exp2f((TEMP_ * (float)lglob / L_) * LOG2E);
  }
#pragma unroll
  for (int i = 0; i < 8; ++i)
    idv[i] = inv_denom[(size_t)b * L_ + lblock + lg * 8 + i];

  v8f acc[4];
  const v8f vzero = {0.f, 0.f, 0.f, 0.f, 0.f, 0.f, 0.f, 0.f};
#pragma unroll
  for (int nt = 0; nt < 4; ++nt) acc[nt] = vzero;

  for (int t0 = 0; t0 < T_; t0 += 32) {
    __syncthreads();   // previous iteration's WMMA reads done before restage

    // ---- stage A: membership tile (computed on the fly, telescoped sigmoid)
    {
      const float ncv = nc_b[t0 + tA];
      const float E   = __builtin_amdgcn_exp2f(-TEMP_ * ncv * LOG2E);
      float slo = __builtin_amdgcn_rcpf(1.f + E * K[0]);
#pragma unroll
      for (int i = 0; i < 8; ++i) {
        const float shi = __builtin_amdgcn_rcpf(1.f + E * K[i + 1]);
        smemA[(lg * 8 + i) * 32 + tA] = (bf16_t)((slo - shi) * idv[i]);
        slo = shi;
      }
    }

    // ---- stage B: x chunk [32 t][128 c] -> LDS transposed [c][t] in bf16
#pragma unroll
    for (int j = 0; j < 4; ++j) {
      const int idx = tid + j * 256;
      const int t   = idx >> 5;
      const int c4  = (idx & 31) * 4;
      const f32x4 xv = *reinterpret_cast<const f32x4*>(
          xb + (size_t)(t0 + t) * C_ + cblock + c4);
      smemB[(c4 + 0) * 32 + t] = (bf16_t)xv.x;
      smemB[(c4 + 1) * 32 + t] = (bf16_t)xv.y;
      smemB[(c4 + 2) * 32 + t] = (bf16_t)xv.z;
      smemB[(c4 + 3) * 32 + t] = (bf16_t)xv.w;
      if (t0 + 32 < T_)   // gfx1250 global_prefetch_b8 for next K chunk
        __builtin_prefetch(xb + (size_t)(t0 + 32 + t) * C_ + cblock + c4, 0, 1);
    }
    __syncthreads();

    // ---- load fragments per ISA 16-bit layouts, issue WMMA
    union Frag { v16bf v; u32x4 u[2]; } a, bfr[4];
    {
      // A 16x32 bf16: lane(half,r): row r, elems 0-7 = K half*8.., 8-15 = K 16+half*8..
      const int row = wr * 16 + n16;
      a.u[0] = *reinterpret_cast<const u32x4*>(&smemA[row * 32 + half * 8]);
      a.u[1] = *reinterpret_cast<const u32x4*>(&smemA[row * 32 + 16 + half * 8]);
    }
#pragma unroll
    for (int nt = 0; nt < 4; ++nt) {
      // B 32x16 bf16: lane(half,n): column n, elems 0-7 = K half*8.., 8-15 = K 16+half*8..
      const int col = wc * 64 + nt * 16 + n16;
      const u32x4* pb = reinterpret_cast<const u32x4*>(&smemB[col * 32 + half * 8]);
      bfr[nt].u[0] = pb[0];
      bfr[nt].u[1] = pb[2];
    }
#pragma unroll
    for (int nt = 0; nt < 4; ++nt) {
      acc[nt] = __builtin_amdgcn_wmma_f32_16x16x32_bf16(
          false, a.v, false, bfr[nt].v, (short)0, acc[nt], false, false);
    }
  }

  // ---- epilogue: C/D layout = VGPR v holds (M = v + half*8, N = n16)
  float* op = out + ((size_t)b * L_ + lblock + wr * 16) * C_ + cblock + wc * 64;
#pragma unroll
  for (int nt = 0; nt < 4; ++nt) {
#pragma unroll
    for (int vv = 0; vv < 8; ++vv) {
      const int M = vv + half * 8;
      op[(size_t)M * C_ + nt * 16 + n16] = acc[nt][vv];
    }
  }
}

// ---------------------------------------------------------------------------
extern "C" void kernel_launch(void* const* d_in, const int* in_sizes, int n_in,
                              void* d_out, int out_size, void* d_ws, size_t ws_size,
                              hipStream_t stream) {
  const float* x    = (const float*)d_in[0];   // (B,T,C) fp32
  const float* w    = (const float*)d_in[1];   // (C,)    fp32
  const float* bias = (const float*)d_in[2];   // (1,)    fp32
  float* out = (float*)d_out;                  // (B,L,C) fp32

  float* ws     = (float*)d_ws;
  float* nc     = ws;                          // B*T floats (scores -> nc in place)
  float* invden = ws + (size_t)B_ * T_;        // B*L floats

  score_kernel <<<(B_ * T_) / 8, 256, 0, stream>>>(x, w, bias, nc);
  cumsum_kernel<<<B_, 1024, 0, stream>>>(nc);
  denom_kernel <<<B_ * L_, 256, 0, stream>>>(nc, invden);

  dim3 g(C_ / 128, L_ / 64, B_);
  pool_gemm_kernel<<<g, 256, 0, stream>>>(x, nc, invden, out);
}
